// Conv_Transformer_7533372637762
// MI455X (gfx1250) — matrix operand order repaired and checked by reading, not compiled, not run on hardware
//
#include <hip/hip_runtime.h>
#include <hip/hip_bf16.h>

// ---------------------------------------------------------------------------
// Conv-Transformer forward for gfx1250 (MI455X).
// All convs run as implicit-GEMM on v_wmma_f32_16x16x32_f16 (f16 in, f32 acc).
// Weights are pre-packed into ISA B-fragment layout; conv kernels do NT WMMAs
// per A-gather (NT = 1, 2 or 4 output tiles per wave).
// ---------------------------------------------------------------------------

typedef __attribute__((ext_vector_type(16))) _Float16 v16h;
typedef __attribute__((ext_vector_type(8)))  float    v8f;

#define Bk  2
#define Lk  8
#define Ck  64
#define Hk  32
#define Wk  32
#define HWp 1024          // pixels per image
#define NEG_SLOPE 0.01f

__device__ __forceinline__ float leakyf(float v) { return v >= 0.f ? v : NEG_SLOPE * v; }

static inline int divUp(int a, int b) { return (a + b - 1) / b; }

// --------------------------- small utility kernels -------------------------

// frames (B,1,H,W,L) -> xh (B*L, 1, H*W) f16
__global__ void k_frames(const float* __restrict__ fr, _Float16* __restrict__ xh) {
  int i = blockIdx.x * 256 + threadIdx.x;
  if (i >= 16 * HWp) return;
  int p = i & 1023, n = i >> 10;
  int b = n >> 3, l = n & 7;
  int h = p >> 5, w = p & 31;
  xh[i] = (_Float16)fr[(((b * Hk + h) * Wk + w) * Lk) + l];
}

// Pack conv weights (Cout, Cin, KHW) f32 into WMMA B-fragment layout f16:
// P[((ntile*nch + chn)*32 + lane)*16 + e], k = chn*32 + (lane>>4)*8 + (e&7) + (e>>3)*16,
// column co = ntile*16 + (lane&15).  Zero-padded K tail.
__global__ void k_pack_b(const float* __restrict__ W, _Float16* __restrict__ P,
                         int Cin, int KHW, int Cout) {
  int Ktot = Cin * KHW;
  int nch = (Ktot + 31) >> 5;
  int total = (Cout >> 4) * nch * 512;
  int i = blockIdx.x * 256 + threadIdx.x;
  if (i >= total) return;
  int e = i & 15;
  int lane = (i >> 4) & 31;
  int rest = i >> 9;                 // ntile*nch + chn
  int chn = rest % nch;
  int ntile = rest / nch;
  int kk = (chn << 5) + ((lane >> 4) << 3) + (e & 7) + ((e >> 3) << 4);
  int co = (ntile << 4) + (lane & 15);
  _Float16 v = (_Float16)0.f;
  if (kk < Ktot) {
    int ci = kk / KHW, r = kk - ci * KHW;
    v = (_Float16)W[(co * Cin + ci) * KHW + r];
  }
  P[i] = v;
}

// ------------------------- templated WMMA implicit-GEMM conv ----------------
// X: (nImages, Cin, 1024) f16.  PB: packed B fragments.  Y: (nImg, Cout, 1024) f32.
// One wave computes a 16(pixels) x (NT*16)(out-channels) tile.
template <int Cin, int KHW, int KW, int R, int NT>
__global__ void k_conv_wmma_t(const _Float16* __restrict__ X,
                              const _Float16* __restrict__ PB,
                              const float* __restrict__ bias,
                              float* __restrict__ Y, int Cout) {
  constexpr int Ktot = Cin * KHW;
  constexpr int nch = (Ktot + 31) / 32;
  int lane = threadIdx.x;
  int mtile = blockIdx.x, ng = blockIdx.y, img = blockIdx.z;
  int lh = lane >> 4, ln = lane & 15;
  int m = (mtile << 4) + ln;         // A-matrix row = pixel index
  int h = m >> 5, w = m & 31;
  const _Float16* __restrict__ Xb = X + ((size_t)img * Cin << 10);
  v8f acc[NT] = {};
  for (int chn = 0; chn < nch; ++chn) {
    int kb = chn << 5;
    v16h a;
#pragma unroll
    for (int e = 0; e < 16; ++e) {
      int kk = kb + lh * 8 + (e & 7) + ((e >= 8) ? 16 : 0);
      _Float16 av = (_Float16)0.f;
      if ((Ktot % 32 == 0) || kk < Ktot) {
        int ci = kk / KHW, r = kk - ci * KHW;     // compile-time constant divisors
        int kh = r / KW, kw = r - kh * KW;
        int ih = h + kh - R, iw = w + kw - R;
        if (R == 0 || (ih >= 0 && ih < Hk && iw >= 0 && iw < Wk))
          av = Xb[(ci << 10) + (ih << 5) + iw];
      }
      a[e] = av;
    }
#pragma unroll
    for (int nt = 0; nt < NT; ++nt) {
      v16h bf = *(const v16h*)(PB +
          ((size_t)(((ng * NT + nt) * nch + chn) * 32 + lane) << 4));
      acc[nt] = __builtin_amdgcn_wmma_f32_16x16x32_f16(false, a, false, bf, (short)0,
                                                       acc[nt], false, false);
    }
  }
#pragma unroll
  for (int nt = 0; nt < NT; ++nt) {
    int co = ((ng * NT + nt) << 4) + ln;
    float bb = bias ? bias[co] : 0.f;
#pragma unroll
    for (int v = 0; v < 8; ++v) {
      int mo = (mtile << 4) + v + lh * 8;
      Y[((size_t)(img * Cout + co) << 10) + mo] = acc[nt][v] + bb;
    }
  }
}

// Pairwise score conv: input = Q[b,i] + K[b,j] formed on the fly. 5x5, 64->64.
// One wave: 16 pixels x all 64 out channels (4 WMMAs per K-chunk, 50 chunks).
__global__ void k_pairconv_wmma(const _Float16* __restrict__ Qh,
                                const _Float16* __restrict__ Kh,
                                const _Float16* __restrict__ PB,
                                const float* __restrict__ bias,
                                float* __restrict__ S) {
  constexpr int nch = 50;                    // 64*25 / 32
  int lane = threadIdx.x;
  int mtile = blockIdx.x, z = blockIdx.z;    // z = (b*8+i)*8+j
  int b = z >> 6, ii = (z >> 3) & 7, jj = z & 7;
  const _Float16* __restrict__ Qb = Qh + ((size_t)((b * 8 + ii) * 64) << 10);
  const _Float16* __restrict__ Kb = Kh + ((size_t)((b * 8 + jj) * 64) << 10);
  int lh = lane >> 4, ln = lane & 15;
  int m = (mtile << 4) + ln;
  int h = m >> 5, w = m & 31;
  v8f acc[4] = {};
  for (int chn = 0; chn < nch; ++chn) {
    int kb = chn << 5;
    v16h a;
#pragma unroll
    for (int e = 0; e < 16; ++e) {
      int kk = kb + lh * 8 + (e & 7) + ((e >= 8) ? 16 : 0);
      int ci = kk / 25, r = kk - ci * 25;
      int kh = r / 5, kw = r - kh * 5;
      int ih = h + kh - 2, iw = w + kw - 2;
      _Float16 av = (_Float16)0.f;
      if (ih >= 0 && ih < Hk && iw >= 0 && iw < Wk) {
        int o = (ci << 10) + (ih << 5) + iw;
        av = (_Float16)(Qb[o] + Kb[o]);
      }
      a[e] = av;
    }
#pragma unroll
    for (int nt = 0; nt < 4; ++nt) {
      v16h bf = *(const v16h*)(PB + ((size_t)((nt * nch + chn) * 32 + lane) << 4));
      acc[nt] = __builtin_amdgcn_wmma_f32_16x16x32_f16(false, a, false, bf, (short)0,
                                                       acc[nt], false, false);
    }
  }
#pragma unroll
  for (int nt = 0; nt < 4; ++nt) {
    int co = (nt << 4) + ln;
    float bb = bias[co];
#pragma unroll
    for (int v = 0; v < 8; ++v) {
      int mo = (mtile << 4) + v + lh * 8;
      S[((size_t)(z * 64 + co) << 10) + mo] = acc[nt][v] + bb;
    }
  }
}

// --------------------------- BatchNorm (per l,c) ----------------------------

__global__ void k_bn_stats(const float* __restrict__ Y, float* __restrict__ mean,
                           float* __restrict__ var, int Cc) {
  int c = blockIdx.x, l = blockIdx.y;
  int tid = threadIdx.x;
  __shared__ float ss[256], sq[256];
  float s = 0.f, q = 0.f;
  for (int idx = tid; idx < 2048; idx += 256) {
    int b = idx >> 10, p = idx & 1023;
    float v = Y[(((b * 8 + l) * Cc + c) << 10) + p];
    s += v; q += v * v;
  }
  ss[tid] = s; sq[tid] = q;
  __syncthreads();
  for (int st = 128; st > 0; st >>= 1) {
    if (tid < st) { ss[tid] += ss[tid + st]; sq[tid] += sq[tid + st]; }
    __syncthreads();
  }
  if (tid == 0) {
    float m = ss[0] / 2048.f;
    mean[l * Cc + c] = m;
    var[l * Cc + c] = sq[0] / 2048.f - m * m;
  }
}

__global__ void k_bn_apply(const float* __restrict__ Y, const float* __restrict__ mean,
                           const float* __restrict__ var, const float* __restrict__ g,
                           const float* __restrict__ be, _Float16* __restrict__ outh,
                           float* __restrict__ outf, int Cc) {
  int i = blockIdx.x * 256 + threadIdx.x;
  int total = 16 * Cc * 1024;
  if (i >= total) return;
  int rest = i >> 10;
  int c = rest % Cc;
  int n = rest / Cc;
  int l = n & 7;
  float m = mean[l * Cc + c], v = var[l * Cc + c];
  float x = (Y[i] - m) * rsqrtf(v + 1e-5f) * g[c] + be[c];
  x = leakyf(x);
  if (outh) outh[i] = (_Float16)x;
  if (outf) outf[i] = x;
}

// --------------------------- positional encoding ---------------------------

__global__ void k_add_pos(float* __restrict__ x) {
  int i = blockIdx.x * 256 + threadIdx.x;
  if (i >= Bk * Lk * Ck * HWp) return;
  int rest = i >> 10;
  int c = rest & 63;
  int l = (rest >> 6) & 7;
  float inv = powf(10000.f, 2.f * (float)(c >> 1) / 64.f);
  float ang = (float)l / inv;
  x[i] += (c & 1) ? cosf(ang) : sinf(ang);
}

// --------------------------- LayerNorm over (C,H,W) ------------------------

__global__ void k_layernorm(const float* __restrict__ X, const float* __restrict__ g,
                            const float* __restrict__ bb, _Float16* __restrict__ outh,
                            float* __restrict__ outf) {
  int n = blockIdx.x;  // b*8+l
  int tid = threadIdx.x;
  const float* xb = X + ((size_t)n << 16);
  __shared__ float ss[256], sq[256];
  __shared__ float mv[2];
  float s = 0.f, q = 0.f;
  for (int i = tid; i < 65536; i += 256) { float v = xb[i]; s += v; q += v * v; }
  ss[tid] = s; sq[tid] = q;
  __syncthreads();
  for (int st = 128; st > 0; st >>= 1) {
    if (tid < st) { ss[tid] += ss[tid + st]; sq[tid] += sq[tid + st]; }
    __syncthreads();
  }
  if (tid == 0) {
    float m = ss[0] / 65536.f;
    mv[0] = m;
    mv[1] = rsqrtf(sq[0] / 65536.f - m * m + 1e-5f);
  }
  __syncthreads();
  float m = mv[0], rs = mv[1];
  for (int i = tid; i < 65536; i += 256) {
    float v = (xb[i] - m) * rs * g[i] + bb[i];
    if (outh) outh[((size_t)n << 16) + i] = (_Float16)v;
    if (outf) outf[((size_t)n << 16) + i] = v;
  }
}

// qkv (n,192,1024) f32 -> Qh/Kh (n,64,1024) f16
__global__ void k_extract_qk(const float* __restrict__ qkv, _Float16* __restrict__ Qh,
                             _Float16* __restrict__ Kh) {
  int i = blockIdx.x * 256 + threadIdx.x;
  if (i >= 16 * 64 * 1024) return;
  int p = i & 1023;
  int c = (i >> 10) & 63;
  int n = i >> 16;
  Qh[i] = (_Float16)qkv[((n * 192 + c) << 10) + p];
  Kh[i] = (_Float16)qkv[((n * 192 + 64 + c) << 10) + p];
}

// softmax over j + weighted V sum + residual
__global__ void k_softmax_av(const float* __restrict__ S, const float* __restrict__ Vb,
                             int vstride, int voff, const float* __restrict__ res,
                             float* __restrict__ out) {
  int idx = blockIdx.x * 256 + threadIdx.x;  // (b,i,c,p)
  if (idx >= Bk * Lk * Ck * HWp) return;
  int p = idx & 1023;
  int c = (idx >> 10) & 63;
  int bi = idx >> 16;  // b*8+i
  int b = bi >> 3;
  float sv[8];
  float mx = -1e30f;
#pragma unroll
  for (int j = 0; j < 8; ++j) {
    float v = S[((size_t)((bi * 8 + j) * 64 + c) << 10) + p];
    sv[j] = v;
    mx = fmaxf(mx, v);
  }
  float den = 0.f;
#pragma unroll
  for (int j = 0; j < 8; ++j) { sv[j] = expf(sv[j] - mx); den += sv[j]; }
  float acc = 0.f;
#pragma unroll
  for (int j = 0; j < 8; ++j) {
    float vv = Vb[((size_t)((b * 8 + j) * vstride + voff + c) << 10) + p];
    acc += sv[j] * vv;
  }
  out[idx] = acc / den + res[idx];
}

// feed-forward over time axis (l: 8 -> 32 -> 8), fused with residual
__global__ void k_ffn(const float* __restrict__ Xn, const float* __restrict__ w1,
                      const float* __restrict__ b1, const float* __restrict__ w2,
                      const float* __restrict__ b2, const float* __restrict__ res,
                      float* __restrict__ out) {
  int idx = blockIdx.x * 256 + threadIdx.x;  // (b,c,p)
  if (idx >= Bk * Ck * HWp) return;
  int p = idx & 1023;
  int c = (idx >> 10) & 63;
  int b = idx >> 16;
  float xt[8];
#pragma unroll
  for (int l = 0; l < 8; ++l) xt[l] = Xn[(((b * 8 + l) * 64 + c) << 10) + p];
  float y[32];
#pragma unroll
  for (int f = 0; f < 32; ++f) {
    float s = b1[f];
#pragma unroll
    for (int l = 0; l < 8; ++l) s += xt[l] * w1[f * 8 + l];
    y[f] = leakyf(s);
  }
#pragma unroll
  for (int l = 0; l < 8; ++l) {
    float s = b2[l];
#pragma unroll
    for (int f = 0; f < 32; ++f) s += y[f] * w2[l * 32 + f];
    int o = (((b * 8 + l) * 64 + c) << 10) + p;
    out[o] = s + res[o];
  }
}

// dec_in[b,l] = enc_in[b,7]
__global__ void k_decin(const float* __restrict__ enc_in, float* __restrict__ dec_in) {
  int i = blockIdx.x * 256 + threadIdx.x;
  if (i >= Bk * Lk * Ck * HWp) return;
  int cp = i & 65535;
  int n = i >> 16;
  int b = n >> 3;
  dec_in[i] = enc_in[((size_t)(b * 8 + 7) << 16) + cp];
}

// pixel head: 1x1 conv C->1, write (B,1,H,W,8)
__global__ void k_pix(const float* __restrict__ X, const float* __restrict__ pw,
                      const float* __restrict__ pb, float* __restrict__ out) {
  int i = blockIdx.x * 256 + threadIdx.x;  // (n=b*8+l, p)
  if (i >= 16 * HWp) return;
  int p = i & 1023;
  int n = i >> 10;
  int b = n >> 3, l = n & 7;
  float s = pb[0];
#pragma unroll
  for (int c = 0; c < 64; ++c) s += X[((n * 64 + c) << 10) + p] * pw[c];
  int h = p >> 5, w = p & 31;
  out[(((b * 32 + h) * 32 + w) << 3) + l] = s;
}

// ----------------------------------- host -----------------------------------

extern "C" void kernel_launch(void* const* d_in, const int* in_sizes, int n_in,
                              void* d_out, int out_size, void* d_ws, size_t ws_size,
                              hipStream_t stream) {
  (void)in_sizes; (void)n_in; (void)out_size; (void)ws_size;

  const float* frames = (const float*)d_in[0];
  const float *cw[5], *cb[5], *cg[5], *cbe[5];
  for (int i = 1; i <= 4; ++i) {
    cw[i]  = (const float*)d_in[1 + (i - 1) * 4 + 0];
    cb[i]  = (const float*)d_in[1 + (i - 1) * 4 + 1];
    cg[i]  = (const float*)d_in[1 + (i - 1) * 4 + 2];
    cbe[i] = (const float*)d_in[1 + (i - 1) * 4 + 3];
  }
  struct LP {
    const float *ln1g, *ln1b, *qkvw, *qkvb, *sw, *sb, *ln2g, *ln2b, *ffw1, *ffb1, *ffw2, *ffb2;
  };
  auto getLP = [&](int base) {
    LP p;
    p.ln1g = (const float*)d_in[base + 0];  p.ln1b = (const float*)d_in[base + 1];
    p.qkvw = (const float*)d_in[base + 2];  p.qkvb = (const float*)d_in[base + 3];
    p.sw   = (const float*)d_in[base + 4];  p.sb   = (const float*)d_in[base + 5];
    p.ln2g = (const float*)d_in[base + 6];  p.ln2b = (const float*)d_in[base + 7];
    p.ffw1 = (const float*)d_in[base + 8];  p.ffb1 = (const float*)d_in[base + 9];
    p.ffw2 = (const float*)d_in[base + 10]; p.ffb2 = (const float*)d_in[base + 11];
    return p;
  };
  LP enc = getLP(17), dec = getLP(29);
  const float* pix_w = (const float*)d_in[41];
  const float* pix_b = (const float*)d_in[42];

  // bump allocator over workspace
  char* base = (char*)d_ws;
  size_t off = 0;
  auto allocB = [&](size_t bytes) -> void* {
    void* p = base + off;
    off = (off + bytes + 255) & ~(size_t)255;
    return p;
  };
  auto allocH = [&](size_t n) { return (_Float16*)allocB(n * 2); };
  auto allocF = [&](size_t n) { return (float*)allocB(n * 4); };

  // packed-B sizes: (Cout/16) * nch * 512 halves
  auto packed_sz = [](int Cin, int KHW, int Cout) {
    int nch = (Cin * KHW + 31) >> 5;
    return (size_t)(Cout >> 4) * nch * 512;
  };
  const int cins[5]  = {0, 1, 16, 32, 64};
  const int couts[5] = {0, 16, 32, 64, 64};
  _Float16* PBc[5];
  for (int i = 1; i <= 4; ++i) PBc[i] = allocH(packed_sz(cins[i], 25, couts[i]));
  _Float16* PQKV_E = allocH(packed_sz(64, 1, 192));
  _Float16* PSW_E  = allocH(packed_sz(64, 25, 64));
  _Float16* PQKV_D = allocH(packed_sz(64, 1, 192));
  _Float16* PSW_D  = allocH(packed_sz(64, 25, 64));

  _Float16* XH_A = allocH((size_t)16 * 64 * 1024);
  _Float16* XH_B = allocH((size_t)16 * 64 * 1024);
  float* CONV32  = allocF((size_t)16 * 192 * 1024);  // conv outputs, later qkv
  float* MEAN = allocF(8 * 64);
  float* VAR  = allocF(8 * 64);
  float* ENC_IN  = allocF((size_t)Bk * Lk * Ck * HWp);
  float* ENC_OUT = allocF((size_t)Bk * Lk * Ck * HWp);
  float* DEC_IN  = allocF((size_t)Bk * Lk * Ck * HWp);
  float* DEC_OUT = allocF((size_t)Bk * Lk * Ck * HWp);
  float* X1   = allocF((size_t)Bk * Lk * Ck * HWp);
  float* LN32 = allocF((size_t)Bk * Lk * Ck * HWp);
  _Float16* LNH = allocH((size_t)16 * 64 * 1024);
  _Float16* QH  = allocH((size_t)16 * 64 * 1024);
  _Float16* KH  = allocH((size_t)16 * 64 * 1024);
  float* S32 = allocF((size_t)Bk * Lk * Lk * Ck * HWp);  // 32 MB scores

  auto pack = [&](const float* W, _Float16* P, int Cin, int KHW, int Cout) {
    int n = (int)packed_sz(Cin, KHW, Cout);
    k_pack_b<<<divUp(n, 256), 256, 0, stream>>>(W, P, Cin, KHW, Cout);
  };
  for (int i = 1; i <= 4; ++i) pack(cw[i], PBc[i], cins[i], 25, couts[i]);
  pack(enc.qkvw, PQKV_E, 64, 1, 192);
  pack(enc.sw, PSW_E, 64, 25, 64);
  pack(dec.qkvw, PQKV_D, 64, 1, 192);
  pack(dec.sw, PSW_D, 64, 25, 64);

  // ---- feature embedding ----
  k_frames<<<divUp(16 * 1024, 256), 256, 0, stream>>>(frames, XH_A);

  auto bn = [&](int i, int Cout, _Float16* outh, float* outf) {
    k_bn_stats<<<dim3(Cout, 8), 256, 0, stream>>>(CONV32, MEAN, VAR, Cout);
    int tot = 16 * Cout * 1024;
    k_bn_apply<<<divUp(tot, 256), 256, 0, stream>>>(CONV32, MEAN, VAR, cg[i], cbe[i],
                                                    outh, outf, Cout);
  };
  // layer 1: 1 -> 16
  k_conv_wmma_t<1, 25, 5, 2, 1><<<dim3(64, 1, 16), 32, 0, stream>>>(
      XH_A, PBc[1], cb[1], CONV32, 16);
  bn(1, 16, XH_B, nullptr);
  // layer 2: 16 -> 32
  k_conv_wmma_t<16, 25, 5, 2, 2><<<dim3(64, 1, 16), 32, 0, stream>>>(
      XH_B, PBc[2], cb[2], CONV32, 32);
  bn(2, 32, XH_A, nullptr);
  // layer 3: 32 -> 64
  k_conv_wmma_t<32, 25, 5, 2, 4><<<dim3(64, 1, 16), 32, 0, stream>>>(
      XH_A, PBc[3], cb[3], CONV32, 64);
  bn(3, 64, XH_B, nullptr);
  // layer 4: 64 -> 64 (writes f32 feat into ENC_IN)
  k_conv_wmma_t<64, 25, 5, 2, 4><<<dim3(64, 1, 16), 32, 0, stream>>>(
      XH_B, PBc[4], cb[4], CONV32, 64);
  bn(4, 64, nullptr, ENC_IN);
  k_add_pos<<<divUp(Bk * Lk * Ck * HWp, 256), 256, 0, stream>>>(ENC_IN);

  // ---- transformer block runner ----
  auto run_block = [&](const float* xin, const float* vbase, int vstride, int voff,
                       const LP& p, const _Float16* pqkv, const _Float16* psw,
                       float* xout) {
    k_layernorm<<<16, 256, 0, stream>>>(xin, p.ln1g, p.ln1b, LNH, nullptr);
    k_conv_wmma_t<64, 1, 1, 0, 4><<<dim3(64, 3, 16), 32, 0, stream>>>(
        LNH, pqkv, p.qkvb, CONV32, 192);
    k_extract_qk<<<divUp(16 * 64 * 1024, 256), 256, 0, stream>>>(CONV32, QH, KH);
    k_pairconv_wmma<<<dim3(64, 1, 128), 32, 0, stream>>>(QH, KH, psw, p.sb, S32);
    const float* vb = vbase ? vbase : CONV32;
    k_softmax_av<<<divUp(Bk * Lk * Ck * HWp, 256), 256, 0, stream>>>(S32, vb, vstride,
                                                                     voff, xin, X1);
    k_layernorm<<<16, 256, 0, stream>>>(X1, p.ln2g, p.ln2b, nullptr, LN32);
    k_ffn<<<divUp(Bk * Ck * HWp, 256), 256, 0, stream>>>(LN32, p.ffw1, p.ffb1, p.ffw2,
                                                         p.ffb2, X1, xout);
  };

  // encoder: self-attention (V from its own qkv, channels 128..191, stride 192)
  run_block(ENC_IN, nullptr, 192, 128, enc, PQKV_E, PSW_E, ENC_OUT);
  // decoder input: repeat last encoder-input timestep
  k_decin<<<divUp(Bk * Lk * Ck * HWp, 256), 256, 0, stream>>>(ENC_IN, DEC_IN);
  // decoder: cross attention, V = enc_out (stride 64, offset 0)
  run_block(DEC_IN, ENC_OUT, 64, 0, dec, PQKV_D, PSW_D, DEC_OUT);
  // pixel head -> d_out (B,1,H,W,8) f32
  k_pix<<<divUp(16 * 1024, 256), 256, 0, stream>>>(DEC_OUT, pix_w, pix_b, (float*)d_out);
}
